// DecoderRNN_14886356648288
// MI455X (gfx1250) — matrix operand order repaired and checked by reading, not compiled, not run on hardware
//
#include <hip/hip_runtime.h>
#include <hip/hip_bf16.h>

// ---------------------------------------------------------------------------
// DecoderRNN (LSTM cell, T=256 steps, B=512, H=1024, O=512) for gfx1250.
// Strategy: split-bf16 (hi+lo) WMMA GEMMs (v_wmma_f32_16x16x32_bf16, 3 per
// tile-k for ~fp32 accuracy), fused cell update, fused softmax. Weights
// converted to split-bf16 once per launch; activations ping-pong in ws.
// ---------------------------------------------------------------------------

typedef __attribute__((ext_vector_type(16))) __bf16 v16bf;
typedef __attribute__((ext_vector_type(8)))  float  v8f;

union Frag16 { v16bf v; uint4 q[2]; };

#define T_STEPS 256
#define BDIM    512
#define HDIM    1024
#define ODIM    512
#define KG      1536   /* O + H : combined gate GEMM K */

__device__ __forceinline__ unsigned short f2bf_rne(float x) {
    unsigned u = __float_as_uint(x);
    unsigned r = u + 0x7fffu + ((u >> 16) & 1u);
    return (unsigned short)(r >> 16);
}
__device__ __forceinline__ float bf2f(unsigned short b) {
    return __uint_as_float(((unsigned)b) << 16);
}
__device__ __forceinline__ void split2(float x, unsigned short& hi, unsigned short& lo) {
    hi = f2bf_rne(x);
    lo = f2bf_rne(x - bf2f(hi));
}
__device__ __forceinline__ float fast_sigmoid(float x) {
    return 1.0f / (1.0f + __expf(-x));
}
__device__ __forceinline__ float fast_tanh(float x) {
    return 1.0f - 2.0f / (__expf(2.0f * x) + 1.0f);
}

// ---------------------------------------------------------------------------
// Weight prep: build combined gate weight [4096 x 1536] = [W_ih | W_hh] and
// W_out [512 x 1024], each as bf16 hi/lo pairs; combined gate bias (fp32).
// ---------------------------------------------------------------------------
__global__ void prep_weights(const float* __restrict__ W_ih,
                             const float* __restrict__ W_hh,
                             const float* __restrict__ b_ih,
                             const float* __restrict__ b_hh,
                             const float* __restrict__ W_out,
                             unsigned short* __restrict__ Wg_hi,
                             unsigned short* __restrict__ Wg_lo,
                             unsigned short* __restrict__ Wo_hi,
                             unsigned short* __restrict__ Wo_lo,
                             float* __restrict__ bg) {
    const long NG = (long)4 * HDIM * KG;        // 6291456
    const long NO = (long)ODIM * HDIM;          // 524288
    const long NB = 4 * HDIM;                   // 4096
    const long NT = NG + NO + NB;
    for (long i = blockIdx.x * (long)blockDim.x + threadIdx.x; i < NT;
         i += (long)gridDim.x * blockDim.x) {
        if (i < NG) {
            long row = i / KG, col = i % KG;
            float w = (col < ODIM) ? W_ih[row * ODIM + col]
                                   : W_hh[row * HDIM + (col - ODIM)];
            unsigned short hi, lo; split2(w, hi, lo);
            Wg_hi[i] = hi; Wg_lo[i] = lo;
        } else if (i < NG + NO) {
            long j = i - NG;
            unsigned short hi, lo; split2(W_out[j], hi, lo);
            Wo_hi[j] = hi; Wo_lo[j] = lo;
        } else {
            long j = i - NG - NO;
            bg[j] = b_ih[j] + b_hh[j];
        }
    }
}

// ---------------------------------------------------------------------------
// State init: A[0] = [ x=0 | h0 ] in split-bf16, c = c0 (fp32 copy).
// ---------------------------------------------------------------------------
__global__ void init_state(const float* __restrict__ h0,
                           const float* __restrict__ c0,
                           unsigned short* __restrict__ A_hi,
                           unsigned short* __restrict__ A_lo,
                           float* __restrict__ c) {
    const long NA = (long)BDIM * KG;
    const long NC = (long)BDIM * HDIM;
    for (long i = blockIdx.x * (long)blockDim.x + threadIdx.x; i < NA + NC;
         i += (long)gridDim.x * blockDim.x) {
        if (i < NA) {
            long row = i / KG, col = i % KG;
            float v = (col < ODIM) ? 0.0f : h0[row * HDIM + (col - ODIM)];
            unsigned short hi, lo; split2(v, hi, lo);
            A_hi[i] = hi; A_lo[i] = lo;
        } else {
            long j = i - NA;
            c[j] = c0[j];
        }
    }
}

// ---------------------------------------------------------------------------
// Kernel G: gates = A @ Wg^T + bg ; LSTM cell update ; write h_{t+1} split-
// bf16 into the h-columns (512..1535) of the OTHER A buffer (ping-pong).
// One wave per (m-tile, h-column-tile); computes all 4 gate tiles (i,f,g,o).
// ---------------------------------------------------------------------------
__global__ __launch_bounds__(256) void lstm_step_gates(
    const unsigned short* __restrict__ Ahi, const unsigned short* __restrict__ Alo,
    const unsigned short* __restrict__ Whi, const unsigned short* __restrict__ Wlo,
    const float* __restrict__ bg, float* __restrict__ c,
    unsigned short* __restrict__ AhiN, unsigned short* __restrict__ AloN) {
    const int lane  = threadIdx.x & 31;
    const int wave  = threadIdx.x >> 5;
    const int gw    = blockIdx.x * 8 + wave;   // 0..2047
    const int mt    = gw & 31;                 // 0..31  (batch tile)
    const int nt    = gw >> 5;                 // 0..63  (H column tile; shared per WG)
    const int row16 = lane & 15;
    const int half  = lane >> 4;
    const int m     = mt * 16 + row16;

    v8f acc0 = {0,0,0,0,0,0,0,0};
    v8f acc1 = acc0, acc2 = acc0, acc3 = acc0;
    v8f acc[4] = {acc0, acc1, acc2, acc3};

    const unsigned short* arh = Ahi + (size_t)m * KG + half * 8;
    const unsigned short* arl = Alo + (size_t)m * KG + half * 8;

    for (int k0 = 0; k0 < KG; k0 += 32) {
        Frag16 ah, al;
        const uint4* pah = (const uint4*)(arh + k0);
        ah.q[0] = pah[0]; ah.q[1] = pah[2];       // K = kb..kb+7, kb+16..kb+23
        const uint4* pal = (const uint4*)(arl + k0);
        al.q[0] = pal[0]; al.q[1] = pal[2];
#pragma unroll
        for (int g = 0; g < 4; ++g) {
            const int n = g * HDIM + nt * 16 + row16;
            Frag16 bh, bl;
            const uint4* pbh = (const uint4*)(Whi + (size_t)n * KG + k0 + half * 8);
            bh.q[0] = pbh[0]; bh.q[1] = pbh[2];   // K = kb..kb+7, kb+16..kb+23
            const uint4* pbl = (const uint4*)(Wlo + (size_t)n * KG + k0 + half * 8);
            bl.q[0] = pbl[0]; bl.q[1] = pbl[2];
            acc[g] = __builtin_amdgcn_wmma_f32_16x16x32_bf16(
                false, ah.v, false, bh.v, (short)0, acc[g], false, false);
            acc[g] = __builtin_amdgcn_wmma_f32_16x16x32_bf16(
                false, ah.v, false, bl.v, (short)0, acc[g], false, false);
            acc[g] = __builtin_amdgcn_wmma_f32_16x16x32_bf16(
                false, al.v, false, bh.v, (short)0, acc[g], false, false);
        }
    }

    const int ncol = nt * 16 + row16;
    const float bi = bg[0 * HDIM + ncol];
    const float bf = bg[1 * HDIM + ncol];
    const float bG = bg[2 * HDIM + ncol];
    const float bo = bg[3 * HDIM + ncol];
#pragma unroll
    for (int r = 0; r < 8; ++r) {
        const int mm = mt * 16 + r + 8 * half;   // C/D layout: VGPR r, halves
        const size_t cidx = (size_t)mm * HDIM + ncol;
        const float cv = c[cidx];
        const float iv = fast_sigmoid(acc[0][r] + bi);
        const float fv = fast_sigmoid(acc[1][r] + bf);
        const float gv = fast_tanh(acc[2][r] + bG);
        const float ov = fast_sigmoid(acc[3][r] + bo);
        const float cn = fv * cv + iv * gv;
        const float hn = ov * fast_tanh(cn);
        c[cidx] = cn;
        unsigned short hi, lo; split2(hn, hi, lo);
        const size_t ai = (size_t)mm * KG + ODIM + ncol;
        AhiN[ai] = hi; AloN[ai] = lo;
    }
}

// ---------------------------------------------------------------------------
// Kernel S: logits = h_new @ W_out^T + b_out ; row softmax over O=512 ;
// write y (fp32, time-reversed slice of d_out) and y as split-bf16 into the
// x-columns (0..511) of the same (new) A buffer for the next step.
// One WG (32 waves) per 16-row strip; wave w owns logit column tile w.
// ---------------------------------------------------------------------------
__global__ __launch_bounds__(1024) void lstm_step_out(
    const unsigned short* __restrict__ Ahi, const unsigned short* __restrict__ Alo,
    const unsigned short* __restrict__ Whi, const unsigned short* __restrict__ Wlo,
    const float* __restrict__ bout, float* __restrict__ y,
    unsigned short* __restrict__ AhiX, unsigned short* __restrict__ AloX) {
    __shared__ float lds[16][ODIM];
    const int lane  = threadIdx.x & 31;
    const int w     = threadIdx.x >> 5;   // 0..31 (column tile)
    const int mt    = blockIdx.x;         // 0..31 (row strip)
    const int row16 = lane & 15;
    const int half  = lane >> 4;
    const int m     = mt * 16 + row16;

    v8f acc = {0,0,0,0,0,0,0,0};
    const unsigned short* arh = Ahi + (size_t)m * KG + ODIM + half * 8;
    const unsigned short* arl = Alo + (size_t)m * KG + ODIM + half * 8;
    const int n = w * 16 + row16;
    const unsigned short* brh = Whi + (size_t)n * HDIM + half * 8;
    const unsigned short* brl = Wlo + (size_t)n * HDIM + half * 8;

    for (int k0 = 0; k0 < HDIM; k0 += 32) {
        Frag16 ah, al, bh, bl;
        const uint4* pah = (const uint4*)(arh + k0);
        ah.q[0] = pah[0]; ah.q[1] = pah[2];
        const uint4* pal = (const uint4*)(arl + k0);
        al.q[0] = pal[0]; al.q[1] = pal[2];
        const uint4* pbh = (const uint4*)(brh + k0);
        bh.q[0] = pbh[0]; bh.q[1] = pbh[2];
        const uint4* pbl = (const uint4*)(brl + k0);
        bl.q[0] = pbl[0]; bl.q[1] = pbl[2];
        acc = __builtin_amdgcn_wmma_f32_16x16x32_bf16(
            false, ah.v, false, bh.v, (short)0, acc, false, false);
        acc = __builtin_amdgcn_wmma_f32_16x16x32_bf16(
            false, ah.v, false, bl.v, (short)0, acc, false, false);
        acc = __builtin_amdgcn_wmma_f32_16x16x32_bf16(
            false, al.v, false, bh.v, (short)0, acc, false, false);
    }

    const float bb = bout[n];
#pragma unroll
    for (int r = 0; r < 8; ++r)
        lds[r + 8 * half][w * 16 + row16] = acc[r] + bb;
    __syncthreads();

    if (w < 16) {
        const int r = w;                       // one wave per strip row
        float mx = -3.402823466e38f;
#pragma unroll
        for (int j = 0; j < 16; ++j) mx = fmaxf(mx, lds[r][lane * 16 + j]);
        for (int s = 16; s >= 1; s >>= 1) mx = fmaxf(mx, __shfl_xor(mx, s, 32));
        float sum = 0.0f;
#pragma unroll
        for (int j = 0; j < 16; ++j) {
            const float e = __expf(lds[r][lane * 16 + j] - mx);
            lds[r][lane * 16 + j] = e;
            sum += e;
        }
        for (int s = 16; s >= 1; s >>= 1) sum += __shfl_xor(sum, s, 32);
        const float inv = 1.0f / sum;
        const int mm = mt * 16 + r;
#pragma unroll
        for (int j = 0; j < 16; ++j) {
            const int col = lane * 16 + j;
            const float val = lds[r][col] * inv;
            y[(size_t)mm * ODIM + col] = val;
            unsigned short hi, lo; split2(val, hi, lo);
            const size_t ai = (size_t)mm * KG + col;
            AhiX[ai] = hi; AloX[ai] = lo;
        }
    }
}

// ---------------------------------------------------------------------------
// Launch
// ---------------------------------------------------------------------------
extern "C" void kernel_launch(void* const* d_in, const int* in_sizes, int n_in,
                              void* d_out, int out_size, void* d_ws, size_t ws_size,
                              hipStream_t stream) {
    const float* h0    = (const float*)d_in[0];  // [1,512,1024]
    const float* c0    = (const float*)d_in[1];  // [1,512,1024]
    const float* W_ih  = (const float*)d_in[2];  // [4096,512]
    const float* W_hh  = (const float*)d_in[3];  // [4096,1024]
    const float* b_ih  = (const float*)d_in[4];  // [4096]
    const float* b_hh  = (const float*)d_in[5];  // [4096]
    const float* W_out = (const float*)d_in[6];  // [512,1024]
    const float* b_out = (const float*)d_in[7];  // [512]
    float* out = (float*)d_out;                  // [256,512,512]

    char* ws = (char*)d_ws;
    size_t off = 0;
    auto take = [&](size_t bytes) -> void* {
        void* p = ws + off;
        off = (off + bytes + 255) & ~(size_t)255;
        return p;
    };
    unsigned short* Wg_hi = (unsigned short*)take((size_t)4 * HDIM * KG * 2);
    unsigned short* Wg_lo = (unsigned short*)take((size_t)4 * HDIM * KG * 2);
    unsigned short* Wo_hi = (unsigned short*)take((size_t)ODIM * HDIM * 2);
    unsigned short* Wo_lo = (unsigned short*)take((size_t)ODIM * HDIM * 2);
    float*          bg    = (float*)take((size_t)4 * HDIM * 4);
    unsigned short* A0_hi = (unsigned short*)take((size_t)BDIM * KG * 2);
    unsigned short* A0_lo = (unsigned short*)take((size_t)BDIM * KG * 2);
    unsigned short* A1_hi = (unsigned short*)take((size_t)BDIM * KG * 2);
    unsigned short* A1_lo = (unsigned short*)take((size_t)BDIM * KG * 2);
    float*          cbuf  = (float*)take((size_t)BDIM * HDIM * 4);

    prep_weights<<<2048, 256, 0, stream>>>(W_ih, W_hh, b_ih, b_hh, W_out,
                                           Wg_hi, Wg_lo, Wo_hi, Wo_lo, bg);
    init_state<<<1024, 256, 0, stream>>>(h0, c0, A0_hi, A0_lo, cbuf);

    unsigned short* Ahi = A0_hi; unsigned short* Alo = A0_lo;
    unsigned short* Bhi = A1_hi; unsigned short* Blo = A1_lo;
    for (int t = 0; t < T_STEPS; ++t) {
        // gates + cell update: read A[p], write h_{t+1} into A[1-p]
        lstm_step_gates<<<256, 256, 0, stream>>>(Ahi, Alo, Wg_hi, Wg_lo, bg,
                                                 cbuf, Bhi, Blo);
        // output projection + softmax: read h from A[1-p], write x into A[1-p]
        float* yslice = out + (size_t)(T_STEPS - 1 - t) * BDIM * ODIM;
        lstm_step_out<<<32, 1024, 0, stream>>>(Bhi, Blo, Wo_hi, Wo_lo, b_out,
                                               yslice, Bhi, Blo);
        // swap ping-pong
        unsigned short* th = Ahi; Ahi = Bhi; Bhi = th;
        unsigned short* tl = Alo; Alo = Blo; Blo = tl;
    }
    (void)in_sizes; (void)n_in; (void)out_size; (void)ws_size;
}
